// Model_23991687315674
// MI455X (gfx1250) — matrix operand order repaired and checked by reading, not compiled, not run on hardware
//
#include <hip/hip_runtime.h>
#include <hip/hip_bf16.h>
#include <math.h>

// ---------------------------------------------------------------------------
// MI455X (gfx1250) implementation.
//  - 5x (3x3 conv 64ch + BN + ReLU) stack via implicit-GEMM bf16 WMMA
//    (V_WMMA_F32_16X16X32_BF16), activations in NHWC bf16.
//  - conv input row-slab staged into LDS by the Tensor Data Mover
//    (TENSOR_LOAD_TO_LDS with iterate mode), waited with s_wait_tensorcnt.
//  - tiny patch CNN branches: LDS-resident direct conv, one wave32 per image.
//  - kNN/top-k/softmax/fold: one workgroup per sample, all in LDS.
//  - conv biases of layers 0..4 omitted: cancelled exactly by batch-norm.
// Input index map (setup_inputs() dict insertion order, leaves depth-first):
//   0:x  1:Wt  2:bt  3..26:metric[m][j]{w,b}  27..34:temp[j]{w,b}
//   35..46:std[j]{w,b}  47..56:bn[j]{g,b}  57:t
// ---------------------------------------------------------------------------

typedef unsigned short u16;
typedef __attribute__((ext_vector_type(16))) __bf16 v16bf;
typedef __attribute__((ext_vector_type(8)))  float  v8f;
typedef __attribute__((ext_vector_type(4)))  unsigned int u32x4;
typedef __attribute__((ext_vector_type(8)))  int i32x8;
typedef __attribute__((ext_vector_type(4)))  int i32x4;

struct Q2 { uint4 a; uint4 b; };   // 32 bytes == one v16bf fragment
static_assert(sizeof(Q2) == 32, "Q2 must be 32B");

__device__ __forceinline__ u16 f2bf(float f) {
    unsigned u = __float_as_uint(f);
    unsigned r = ((u >> 16) & 1u) + 0x7FFFu;      // round-to-nearest-even
    return (u16)((u + r) >> 16);
}
__device__ __forceinline__ float bf2f(u16 h) {
    return __uint_as_float(((unsigned)h) << 16);
}

// ---------------------------------------------------------------------------
// TDM: iterate-mode row-slab load.  Copies `nrows` contiguous global rows of
// `rowUnits` dwords each into LDS at stride `ldsIncBytes`.
// D# packing per CDNA5 ISA (group0/1/2).
// This toolchain's builtin is the 6-arg (clang-23 / therock-10.0) flavor.
// ---------------------------------------------------------------------------
__device__ __forceinline__ void tdm_load_rows(unsigned ldsAddr, unsigned long long gAddr,
                                              unsigned rowUnits, unsigned nrows,
                                              unsigned ldsIncBytes) {
    u32x4 g0;
    g0[0] = 1u;                                            // count=1, user D#
    g0[1] = ldsAddr;                                       // lds_addr (bytes)
    g0[2] = (unsigned)(gAddr & 0xFFFFFFFFull);             // global_addr[31:0]
    g0[3] = (unsigned)((gAddr >> 32) & 0x1FFFFFFull)       // global_addr[56:32]
          | 0x80000000u;                                   // type = 2 ("image")
    i32x8 g1;
    g1[0] = (2 << 16) | (1 << 19);                         // data_size=4B, iterate_enable
    g1[1] = (int)((rowUnits & 0xFFFFu) << 16);             // tensor_dim0[15:0]
    g1[2] = (int)((rowUnits >> 16) | (1u << 16));          // tensor_dim0[31:16] | tensor_dim1=1
    g1[3] = (int)((rowUnits & 0xFFFFu) << 16);             // tile_dim0
    g1[4] = 1;                                             // tile_dim1=1, tile_dim2=0
    g1[5] = (int)rowUnits;                                 // tensor_dim0_stride[31:0]
    g1[6] = 0;
    g1[7] = 0;
    i32x4 g2;
    g2[0] = 0;                                             // tensor_dim2
    g2[1] = (int)(ldsIncBytes >> 2);                       // lds_addr_increment (dword units)
    g2[2] = (int)rowUnits;                                 // global_addr_increment[31:0]
    g2[3] = (int)((nrows - 1u) << 16);                     // iterate_count
    i32x4 g3 = {0, 0, 0, 0};
    i32x8 g4 = {0, 0, 0, 0, 0, 0, 0, 0};
    __builtin_amdgcn_tensor_load_to_lds(g0, g1, g2, g3, g4, 0);
}

// ---------------------------------------------------------------------------
// Kernel 1: x += one_hot(t) @ Wt + bt   (row gather of Wt)
// ---------------------------------------------------------------------------
__global__ void k_temb(const float* __restrict__ x, const float* __restrict__ Wt,
                       const float* __restrict__ bt, const int* __restrict__ t,
                       float* __restrict__ xa) {
    size_t i = (size_t)blockIdx.x * 256 + threadIdx.x;     // 512*3072
    if (i >= (size_t)512 * 3072) return;
    int b = (int)(i / 3072), e = (int)(i % 3072);
    xa[i] = x[i] + Wt[(size_t)t[b] * 3072 + e] + bt[e];
}

// ---------------------------------------------------------------------------
// Kernel 2: unfold into patches[B][49][3][8][8]
// ---------------------------------------------------------------------------
__global__ void k_patch(const float* __restrict__ xa, float* __restrict__ patches) {
    size_t i = (size_t)blockIdx.x * 256 + threadIdx.x;     // 512*49*192
    if (i >= (size_t)512 * 49 * 192) return;
    int b = (int)(i / (49 * 192));
    int r = (int)(i % (49 * 192));
    int l = r / 192, q = r % 192;
    int c = q / 64, pp = q % 64, py = pp / 8, px = pp % 8;
    int gi = l / 7, gj = l % 7;
    patches[i] = xa[((size_t)(b * 3 + c) * 32 + (gi * 4 + py)) * 32 + (gj * 4 + px)];
}

// ---------------------------------------------------------------------------
// Kernel 3: one tiny CNN branch over 25088 patch-images.
// One wave32 per image, 8 images per workgroup, intermediates + small weights
// in LDS (conv4 weights streamed from L2).
// ---------------------------------------------------------------------------
__global__ void k_branch(const float* __restrict__ patches,
                         const float* __restrict__ w1, const float* __restrict__ b1,
                         const float* __restrict__ w2, const float* __restrict__ b2,
                         const float* __restrict__ w3, const float* __restrict__ b3,
                         const float* __restrict__ w4, const float* __restrict__ b4,
                         int outDim, int doSoftplus, float* __restrict__ outp) {
    __shared__ float sW1[270], sB1[10], sW2[1800], sB2[20], sW3[5400], sB3[30];
    __shared__ float sImg[8][192], sH1[8][360], sH2[8][320], sH3[8][120];

    int tid = threadIdx.x;
    for (int i = tid; i < 270;  i += 256) sW1[i] = w1[i];
    for (int i = tid; i < 1800; i += 256) sW2[i] = w2[i];
    for (int i = tid; i < 5400; i += 256) sW3[i] = w3[i];
    if (tid < 10) sB1[tid] = b1[tid];
    if (tid < 20) sB2[tid] = b2[tid];
    if (tid < 30) sB3[tid] = b3[tid];

    int wv = tid >> 5, lane = tid & 31;
    int img = blockIdx.x * 8 + wv;                 // < 25088 always
    for (int i = lane; i < 192; i += 32) sImg[wv][i] = patches[(size_t)img * 192 + i];
    __syncthreads();

    // conv1: 3->10, 8x8 -> 6x6, relu
    for (int o = lane; o < 360; o += 32) {
        int oc = o / 36, p = o % 36, oy = p / 6, ox = p % 6;
        float a = sB1[oc];
        #pragma unroll
        for (int ic = 0; ic < 3; ++ic)
            #pragma unroll
            for (int ky = 0; ky < 3; ++ky)
                #pragma unroll
                for (int kx = 0; kx < 3; ++kx)
                    a += sImg[wv][ic * 64 + (oy + ky) * 8 + (ox + kx)] *
                         sW1[((oc * 3 + ic) * 3 + ky) * 3 + kx];
        sH1[wv][oc * 36 + p] = fmaxf(a, 0.f);
    }
    __syncthreads();
    // conv2: 10->20, 6x6 -> 4x4, relu
    for (int o = lane; o < 320; o += 32) {
        int oc = o / 16, p = o % 16, oy = p / 4, ox = p % 4;
        float a = sB2[oc];
        for (int ic = 0; ic < 10; ++ic)
            #pragma unroll
            for (int ky = 0; ky < 3; ++ky)
                #pragma unroll
                for (int kx = 0; kx < 3; ++kx)
                    a += sH1[wv][ic * 36 + (oy + ky) * 6 + (ox + kx)] *
                         sW2[((oc * 10 + ic) * 3 + ky) * 3 + kx];
        sH2[wv][oc * 16 + p] = fmaxf(a, 0.f);
    }
    __syncthreads();
    // conv3: 20->30, 4x4 -> 2x2, relu
    for (int o = lane; o < 120; o += 32) {
        int oc = o / 4, p = o % 4, oy = p / 2, ox = p % 2;
        float a = sB3[oc];
        for (int ic = 0; ic < 20; ++ic)
            #pragma unroll
            for (int ky = 0; ky < 3; ++ky)
                #pragma unroll
                for (int kx = 0; kx < 3; ++kx)
                    a += sH2[wv][ic * 16 + (oy + ky) * 4 + (ox + kx)] *
                         sW3[((oc * 20 + ic) * 3 + ky) * 3 + kx];
        sH3[wv][oc * 4 + p] = fmaxf(a, 0.f);
    }
    __syncthreads();
    // conv4: 30->outDim, 2x2 kernel stride 4 -> 1x1 (no relu)
    for (int oc = lane; oc < outDim; oc += 32) {
        float a = b4[oc];
        for (int ic = 0; ic < 30; ++ic)
            #pragma unroll
            for (int ky = 0; ky < 2; ++ky)
                #pragma unroll
                for (int kx = 0; kx < 2; ++kx)
                    a += sH3[wv][ic * 4 + ky * 2 + kx] *
                         w4[((oc * 30 + ic) * 2 + ky) * 2 + kx];
        if (doSoftplus) {
            a = (a > 20.f) ? a : log1pf(__expf(a));
            a += 0.001f;
        }
        outp[(size_t)img * outDim + oc] = a;
    }
}

// ---------------------------------------------------------------------------
// Kernel 4: per-sample kNN (3 metrics) + softmax weights + gather-fold.
// One workgroup per b. Writes bf16 NHWC activation [b][y][x][96] (f>=66 -> 0).
// ---------------------------------------------------------------------------
__global__ void k_knn(const float* __restrict__ emb, const float* __restrict__ tempb,
                      const float* __restrict__ patches, u16* __restrict__ act0) {
    __shared__ float sEmb[3 * 49 * 40];
    __shared__ float sSq[147];
    __shared__ float sTemp[147];       // [l*3 + m]
    __shared__ int   sIdx[3 * 49 * 7];
    __shared__ float sW[3 * 49 * 7];

    int bId = blockIdx.x, tid = threadIdx.x;
    for (int i = tid; i < 5880; i += 256) {
        int m = i / 1960, rem = i % 1960, l = rem / 40, d = rem % 40;
        sEmb[i] = emb[(size_t)m * 25088 * 40 + ((size_t)bId * 49 + l) * 40 + d];
    }
    for (int i = tid; i < 147; i += 256)
        sTemp[i] = tempb[(size_t)bId * 147 + i];
    __syncthreads();
    if (tid < 147) {
        int m = tid / 49, l = tid % 49;
        float s = 0.f;
        #pragma unroll
        for (int d = 0; d < 40; ++d) { float e = sEmb[(m * 49 + l) * 40 + d]; s += e * e; }
        sSq[tid] = s;
    }
    __syncthreads();
    if (tid < 147) {
        int m = tid / 49, l = tid % 49;
        float e[40];
        #pragma unroll
        for (int d = 0; d < 40; ++d) e[d] = sEmb[(m * 49 + l) * 40 + d];
        float bd[7]; int bi[7];
        #pragma unroll
        for (int k = 0; k < 7; ++k) { bd[k] = 3.0e38f; bi[k] = 0; }
        float sqL = sSq[tid];
        for (int j = 0; j < 49; ++j) {
            float dot = 0.f;
            #pragma unroll
            for (int d = 0; d < 40; ++d) dot += e[d] * sEmb[(m * 49 + j) * 40 + d];
            float d2 = sqL + sSq[m * 49 + j] - 2.f * dot;
            if (j == l) d2 += 1.0e9f;
            if (d2 < bd[6]) {                        // sorted insert (ascending d2)
                int k = 6;
                while (k > 0 && bd[k - 1] > d2) { bd[k] = bd[k - 1]; bi[k] = bi[k - 1]; --k; }
                bd[k] = d2; bi[k] = j;
            }
        }
        float t = sTemp[l * 3 + m];
        float m0 = -bd[0] / t, se = 0.f, wv[7];
        #pragma unroll
        for (int k = 0; k < 7; ++k) { wv[k] = __expf((-bd[k] / t) - m0); se += wv[k]; }
        float inv = 1.f / se;
        #pragma unroll
        for (int k = 0; k < 7; ++k) {
            sIdx[(m * 49 + l) * 7 + k] = bi[k];
            sW [(m * 49 + l) * 7 + k] = wv[k] * inv;
        }
    }
    __syncthreads();
    // gather-fold: each output pixel covered by <=4 patches; divisor == count.
    const float* pfb = patches + (size_t)bId * 49 * 192;
    for (int v = tid; v < 1024 * 96; v += 256) {
        int p = v / 96, f = v % 96;
        int y = p >> 5, xq = p & 31;
        u16 hv = 0;
        if (f < 66) {
            int g = f / 3, ch = f % 3;
            int gy0 = y >> 2, gx0 = xq >> 2;
            float sum = 0.f; int cnt = 0;
            #pragma unroll
            for (int a = 0; a < 2; ++a) {
                int gy = gy0 - a; if (gy < 0 || gy > 6) continue;
                int py = y - gy * 4;
                #pragma unroll
                for (int c2 = 0; c2 < 2; ++c2) {
                    int gx = gx0 - c2; if (gx < 0 || gx > 6) continue;
                    int px = xq - gx * 4;
                    ++cnt;
                    int l = gy * 7 + gx;
                    int pix = ch * 64 + py * 8 + px;
                    float val;
                    if (g == 0) {
                        val = pfb[l * 192 + pix];
                    } else {
                        int m = (g - 1) / 7, k = (g - 1) % 7;
                        int jj = sIdx[(m * 49 + l) * 7 + k];
                        val = pfb[jj * 192 + pix] * sW[(m * 49 + l) * 7 + k];
                    }
                    sum += val;
                }
            }
            hv = f2bf(sum / (float)cnt);
        }
        act0[((size_t)(bId * 32 + y) * 32 + xq) * 96 + f] = hv;
    }
}

// ---------------------------------------------------------------------------
// Kernel 5: weight prep OIHW fp32 -> [O=64][ky][kx][CP] bf16, zero-pad C.
// ---------------------------------------------------------------------------
__global__ void k_wprep(const float* __restrict__ w, u16* __restrict__ wbf,
                        int I, int CP) {
    int i = blockIdx.x * 256 + threadIdx.x;          // 64*9*CP
    if (i >= 64 * 9 * CP) return;
    int o = i / (9 * CP), r = i % (9 * CP), tap = r / CP, c = r % CP;
    int ky = tap / 3, kx = tap % 3;
    float v = (c < I) ? w[((size_t)(o * I + c) * 3 + ky) * 3 + kx] : 0.f;
    wbf[i] = f2bf(v);
}

// ---------------------------------------------------------------------------
// Kernel 6: implicit-GEMM 3x3 SAME conv via V_WMMA_F32_16X16X32_BF16.
// One workgroup per (b, y): 8 waves = 2 M-tiles(16 px) x 4 N-tiles(16 ch).
// A slab (3 rows x 34 padded cols x CP) staged in LDS by the TDM; the pad
// columns and out-of-image rows are VALU-zeroed (disjoint LDS regions, no
// ordering hazard vs the DMA). B streamed from L2.
// ---------------------------------------------------------------------------
template <int CP>
__global__ void k_conv(const u16* __restrict__ in, const u16* __restrict__ wt,
                       float* __restrict__ out) {
    __shared__ u16 sIn[3 * 34 * CP];
    int bh = blockIdx.x;
    int b = bh >> 5, y = bh & 31;
    int tid = threadIdx.x;

    __builtin_prefetch(wt, 0, 3);                   // global_prefetch_b8

    // zero pad columns (xp = 0, 33) for all 3 slab rows
    for (int i = tid; i < 3 * CP; i += 256) {
        int ry = i / CP, c = i % CP;
        sIn[(ry * 34 + 0)  * CP + c] = 0;
        sIn[(ry * 34 + 33) * CP + c] = 0;
    }
    // zero out-of-image edge rows
    if (y == 0)  for (int i = tid; i < 32 * CP; i += 256) sIn[(0 * 34 + 1) * CP + i] = 0;
    if (y == 31) for (int i = tid; i < 32 * CP; i += 256) sIn[(2 * 34 + 1) * CP + i] = 0;

    // TDM: one iterate-mode descriptor moves the valid rows (uniform per WG).
    if (tid < 32) {                                  // wave 0 only
        int gy0 = (y == 0) ? 0 : y - 1;
        int gy1 = (y == 31) ? 31 : y + 1;
        unsigned nrows = (unsigned)(gy1 - gy0 + 1);
        int ryStart = (y == 0) ? 1 : 0;
        unsigned ldsAddr = (unsigned)(unsigned long long)(void*)&sIn[(ryStart * 34 + 1) * CP];
        unsigned long long gAddr =
            (unsigned long long)(const void*)(in + ((size_t)(b * 32 + gy0) * 32) * CP);
        tdm_load_rows(ldsAddr, gAddr, (32u * CP * 2u) >> 2, nrows, 34u * CP * 2u);
        __builtin_amdgcn_s_wait_tensorcnt(0);
    }
    __syncthreads();

    int wv = tid >> 5, lane = tid & 31;
    int mt = wv & 1, nt = wv >> 1;
    int Mrow = lane & 15, hi = lane >> 4;
    int n = nt * 16 + Mrow;

    v8f acc = {};
    #pragma unroll
    for (int ty = 0; ty < 3; ++ty)
        #pragma unroll
        for (int tx = 0; tx < 3; ++tx)
            #pragma unroll
            for (int kc = 0; kc < CP / 32; ++kc) {
                // A fragment: 16x32 bf16, rows = output x positions.
                int xp = mt * 16 + Mrow + tx;        // padded coords
                int abase = (ty * 34 + xp) * CP + kc * 32 + hi * 8;
                Q2 qa;
                qa.a = *(const uint4*)&sIn[abase];
                qa.b = *(const uint4*)&sIn[abase + 16];
                v16bf A = __builtin_bit_cast(v16bf, qa);
                // B fragment: 32x16 bf16, col = out channel (K contiguous).
                const u16* wr = wt + ((size_t)n * 9 + (ty * 3 + tx)) * CP + kc * 32 + hi * 8;
                Q2 qb;
                qb.a = *(const uint4*)wr;
                qb.b = *(const uint4*)(wr + 16);
                v16bf Bm = __builtin_bit_cast(v16bf, qb);
                acc = __builtin_amdgcn_wmma_f32_16x16x32_bf16(
                        false, A, false, Bm, (short)0, acc, false, false);
            }

    size_t rowbase = (size_t)(b * 32 + y) * 32;
    #pragma unroll
    for (int r = 0; r < 8; ++r) {
        int Mr = r + hi * 8;
        out[(rowbase + mt * 16 + Mr) * 64 + nt * 16 + Mrow] = acc[r];
    }
}

// ---------------------------------------------------------------------------
// Kernel 7: per-channel sum / sumsq over [B*H*W][64] (for batch-norm).
// ---------------------------------------------------------------------------
__global__ void k_stats(const float* __restrict__ cv, float* __restrict__ stats) {
    __shared__ float red[256];
    int tid = threadIdx.x;
    int c = tid & 63, rg = tid >> 6;
    float s = 0.f, s2 = 0.f;
    size_t base = (size_t)blockIdx.x * 1024;
    for (int rr = 0; rr < 256; ++rr) {
        size_t r = base + (size_t)rr * 4 + rg;
        float v = cv[r * 64 + c];
        s += v; s2 += v * v;
    }
    red[tid] = s;  __syncthreads();
    if (tid < 64) atomicAdd(&stats[tid], red[tid] + red[tid + 64] + red[tid + 128] + red[tid + 192]);
    __syncthreads();
    red[tid] = s2; __syncthreads();
    if (tid < 64) atomicAdd(&stats[64 + tid], red[tid] + red[tid + 64] + red[tid + 128] + red[tid + 192]);
}

// ---------------------------------------------------------------------------
// Kernel 8: BN + ReLU, fp32 -> bf16 NHWC activation for next conv.
// ---------------------------------------------------------------------------
__global__ void k_bnrelu(const float* __restrict__ cv, const float* __restrict__ stats,
                         const float* __restrict__ g, const float* __restrict__ bb,
                         u16* __restrict__ outa) {
    size_t i = (size_t)blockIdx.x * 256 + threadIdx.x;    // 512*1024*64
    if (i >= (size_t)512 * 1024 * 64) return;
    int c = (int)(i & 63);
    const float N = 524288.f;
    float mean = stats[c] / N;
    float var  = stats[64 + c] / N - mean * mean;
    float v = (cv[i] - mean) * rsqrtf(var + 1e-5f) * g[c] + bb[c];
    outa[i] = f2bf(fmaxf(v, 0.f));
}

// ---------------------------------------------------------------------------
// Kernel 9: final 3x3 conv 64 -> 3, writes NCHW fp32 output.
// ---------------------------------------------------------------------------
__global__ void k_final(const u16* __restrict__ act, const float* __restrict__ w,
                        const float* __restrict__ bias, float* __restrict__ out) {
    __shared__ float sw[1728];
    __shared__ float sb[3];
    for (int i = threadIdx.x; i < 1728; i += 256) sw[i] = w[i];
    if (threadIdx.x < 3) sb[threadIdx.x] = bias[threadIdx.x];
    __syncthreads();
    int p = blockIdx.x * 256 + threadIdx.x;               // 512*1024
    int b = p >> 10, y = (p >> 5) & 31, x = p & 31;
    float a0 = sb[0], a1 = sb[1], a2 = sb[2];
    #pragma unroll
    for (int ky = 0; ky < 3; ++ky) {
        int gy = y + ky - 1; if (gy < 0 || gy > 31) continue;
        #pragma unroll
        for (int kx = 0; kx < 3; ++kx) {
            int gx = x + kx - 1; if (gx < 0 || gx > 31) continue;
            const u16* ip = act + (((size_t)(b * 32 + gy) * 32) + gx) * 64;
            for (int c = 0; c < 64; ++c) {
                float av = bf2f(ip[c]);
                a0 += av * sw[((0 * 64 + c) * 3 + ky) * 3 + kx];
                a1 += av * sw[((1 * 64 + c) * 3 + ky) * 3 + kx];
                a2 += av * sw[((2 * 64 + c) * 3 + ky) * 3 + kx];
            }
        }
    }
    size_t ob = ((size_t)b * 3) * 1024 + (size_t)y * 32 + x;
    out[ob]          = a0;
    out[ob + 1024]   = a1;
    out[ob + 2048]   = a2;
}

// ---------------------------------------------------------------------------
// Host orchestration
// ---------------------------------------------------------------------------
extern "C" void kernel_launch(void* const* d_in, const int* in_sizes, int n_in,
                              void* d_out, int out_size, void* d_ws, size_t ws_size,
                              hipStream_t stream) {
    (void)in_sizes; (void)n_in; (void)out_size; (void)ws_size;

    const float* x  = (const float*)d_in[0];
    const float* Wt = (const float*)d_in[1];
    const float* bt = (const float*)d_in[2];
    const int*   tt = (const int*)d_in[57];
    auto mw = [&](int m, int j) { return (const float*)d_in[3 + (m * 4 + j) * 2]; };
    auto mb = [&](int m, int j) { return (const float*)d_in[4 + (m * 4 + j) * 2]; };
    auto tw = [&](int j) { return (const float*)d_in[27 + 2 * j]; };
    auto tb = [&](int j) { return (const float*)d_in[28 + 2 * j]; };
    auto sw = [&](int j) { return (const float*)d_in[35 + 2 * j]; };
    auto sb = [&](int j) { return (const float*)d_in[36 + 2 * j]; };
    auto bg = [&](int j) { return (const float*)d_in[47 + 2 * j]; };
    auto bb = [&](int j) { return (const float*)d_in[48 + 2 * j]; };

    char* base = (char*)d_ws;
    size_t off = 0;
    auto carve = [&](size_t bytes) -> char* {
        char* p = base + off;
        off = (off + bytes + 255) & ~(size_t)255;
        return p;
    };
    float* xa      = (float*)carve((size_t)512 * 3072 * 4);
    float* patches = (float*)carve((size_t)512 * 49 * 192 * 4);
    float* tempb   = (float*)carve((size_t)512 * 49 * 3 * 4);
    float* emb     = (float*)carve((size_t)3 * 25088 * 40 * 4);
    u16*   wbf0    = (u16*)  carve((size_t)64 * 9 * 96 * 2);
    u16*   wbf[4];
    for (int i = 0; i < 4; ++i) wbf[i] = (u16*)carve((size_t)64 * 9 * 64 * 2);
    float* stats   = (float*)carve(128 * 4);
    u16*   bufA    = (u16*)  carve((size_t)512 * 1024 * 96 * 2);   // 96ch then 64ch reuse
    u16*   bufB    = (u16*)  carve((size_t)512 * 1024 * 64 * 2);
    float* cvout   = (float*)carve((size_t)512 * 1024 * 64 * 4);

    // 1) time-embedding add, 2) unfold
    k_temb <<<6144,  256, 0, stream>>>(x, Wt, bt, tt, xa);
    k_patch<<<18816, 256, 0, stream>>>(xa, patches);

    // 3) branches: 3 metric embeddings (40-d) + temperature (softplus)
    for (int m = 0; m < 3; ++m)
        k_branch<<<3136, 256, 0, stream>>>(patches,
            mw(m, 0), mb(m, 0), mw(m, 1), mb(m, 1),
            mw(m, 2), mb(m, 2), mw(m, 3), mb(m, 3),
            40, 0, emb + (size_t)m * 25088 * 40);
    k_branch<<<3136, 256, 0, stream>>>(patches,
        tw(0), tb(0), tw(1), tb(1), tw(2), tb(2), tw(3), tb(3),
        3, 1, tempb);

    // 4) kNN + softmax gather + fold -> bf16 NHWC (C padded to 96)
    k_knn<<<512, 256, 0, stream>>>(emb, tempb, patches, bufA);

    // 5) weight prep for the WMMA conv stack
    k_wprep<<<216, 256, 0, stream>>>(sw(0), wbf0, 66, 96);
    for (int i = 1; i <= 4; ++i)
        k_wprep<<<144, 256, 0, stream>>>(sw(i), wbf[i - 1], 64, 64);

    // 6) conv -> stats -> BN+ReLU, x5 (ping-pong bf16 activations)
    u16* cin = bufA;
    u16* cnext = bufB;
    for (int i = 0; i < 5; ++i) {
        (void)hipMemsetAsync(stats, 0, 512, stream);
        if (i == 0)
            k_conv<96><<<16384, 256, 0, stream>>>(cin, wbf0, cvout);
        else
            k_conv<64><<<16384, 256, 0, stream>>>(cin, wbf[i - 1], cvout);
        k_stats <<<512,    256, 0, stream>>>(cvout, stats);
        k_bnrelu<<<131072, 256, 0, stream>>>(cvout, stats, bg(i), bb(i), cnext);
        u16* t2 = cin; cin = cnext; cnext = t2;
    }

    // 7) final 64 -> 3 conv, NCHW fp32 output
    k_final<<<2048, 256, 0, stream>>>(cin, sw(5), sb(5), (float*)d_out);
}
